// MyGNN_43654047596744
// MI455X (gfx1250) — compile-verified, matrix operand order checked by reading, not run on hardware
//
#include <hip/hip_runtime.h>

typedef __attribute__((ext_vector_type(16))) _Float16 v16h;
typedef __attribute__((ext_vector_type(8)))  float    v8f;
typedef int v4i __attribute__((vector_size(16)));

#define HID 64
#define NUM_LAYERS 3
#define NUM_GRAPHS 512
#define NUM_CLASSES 6
#define BN_EPS 1e-5f

#if defined(__has_builtin)
#  if __has_builtin(__builtin_amdgcn_global_load_async_to_lds_b128)
#    define HAVE_ASYNC_LDS 1
#  endif
#endif

__device__ __forceinline__ void wait_async_lds() {
#if defined(HAVE_ASYNC_LDS)
#  if __has_builtin(__builtin_amdgcn_s_wait_asynccnt)
  __builtin_amdgcn_s_wait_asynccnt(0);
#  else
  asm volatile("s_wait_asynccnt 0" ::: "memory");
#  endif
#endif
}

// ---------- utility kernels ----------
__global__ void fill_f32(float* p, int n, float v) {
  int i = blockIdx.x * blockDim.x + threadIdx.x;
  if (i < n) p[i] = v;
}

__global__ void deg_kernel(const int* __restrict__ dst, int E, float* __restrict__ deg) {
  int e = blockIdx.x * blockDim.x + threadIdx.x;
  if (e < E) atomicAdd(&deg[dst[e]], 1.0f);
}

__global__ void rsqrt_kernel(float* deg, int n) {
  int i = blockIdx.x * blockDim.x + threadIdx.x;
  if (i < n) deg[i] = rsqrtf(deg[i]);
}

// Convert W[K,64] f32 (row-major) into f16 pre-swizzled B-fragment layout:
// chunk g = (kb*64 + n)*2 + h holds 16 halves j=0..15 with
//   K = kb*32 + h*8 + (j<8 ? j : j+8), column n.
// Flat half index i = g*16 + j.
__global__ void cvt_swz_kernel(const float* __restrict__ W, _Float16* __restrict__ o, int K) {
  int i = blockIdx.x * blockDim.x + threadIdx.x;
  if (i >= K * HID) return;
  int j  = i & 15;
  int h  = (i >> 4) & 1;
  int n  = (i >> 5) & 63;
  int kb = i >> 11;
  int k  = kb * 32 + h * 8 + (j < 8 ? j : j + 8);
  o[i] = (_Float16)W[k * HID + n];
}

// ---------- WMMA GEMM: out[N,64] = A[N,K] @ W[K,64] (+bias, relu) ----------
// Block = 256 threads = 8 waves; block covers 32 rows x 64 cols -> 8 16x16 tiles.
// B: pre-swizzled f16 weights async-copied global->LDS (ASYNCcnt).
// A: 32xK f32 tile staged+converted+swizzled into LDS cooperatively.
// Inner loop: one v16h LDS load for A, one for B, one v_wmma per 32-K step.
template <int K, bool BIAS, bool RELU, bool GUARD>
__global__ void gemm_wmma(const float* __restrict__ A, const _Float16* __restrict__ Wswz,
                          const float* __restrict__ bias, float* __restrict__ out,
                          int Nrows, int row0) {
  constexpr int KB = K / 32;
  __shared__ v16h Bswz[KB * 64 * 2];   // [(kb*64 + n)*2 + h]
  __shared__ v16h Aswz[2 * KB * 32];   // [((t*KB + kb)*32 + lane)]

  // --- stage B: verbatim async copy of swizzled weights ---
#if defined(HAVE_ASYNC_LDS)
  {
    const char* gw = (const char*)Wswz;
    char*       lw = (char*)Bswz;
#pragma unroll 2
    for (int i = threadIdx.x * 16; i < KB * 64 * 2 * 32; i += blockDim.x * 16) {
      __builtin_amdgcn_global_load_async_to_lds_b128(
          (__attribute__((address_space(1))) v4i*)(gw + i),
          (__attribute__((address_space(3))) v4i*)(lw + i),
          0, 0);
    }
  }
#else
  {
    const _Float16* gw = (const _Float16*)Wswz;
    _Float16*       lw = (_Float16*)Bswz;
    for (int i = threadIdx.x; i < KB * 64 * 2 * 16; i += blockDim.x) lw[i] = gw[i];
  }
#endif

  // --- stage A: load f32 tile, convert, store fragment-swizzled ---
  const int blockrow = row0 + blockIdx.x * 32;
  for (int ch = threadIdx.x; ch < 2 * KB * 32; ch += blockDim.x) {
    int l    = ch & 31;
    int rest = ch >> 5;           // t*KB + kb
    int kb   = rest % KB;
    int t    = rest / KB;
    int row  = blockrow + t * 16 + (l & 15);
    if (GUARD && row >= Nrows) row = Nrows - 1;
    const float* __restrict__ ar = A + (long)row * K + kb * 32 + (l >> 4) * 8;
    v16h av;
#pragma unroll
    for (int j = 0; j < 8; j++) av[j]     = (_Float16)ar[j];
#pragma unroll
    for (int j = 0; j < 8; j++) av[8 + j] = (_Float16)ar[16 + j];
    Aswz[ch] = av;
  }

  wait_async_lds();
  __syncthreads();

  // --- compute ---
  const int lane = threadIdx.x & 31;
  const int wave = threadIdx.x >> 5;
  const int t    = wave >> 2;               // row tile 0/1
  const int n    = (wave & 3) * 16 + (lane & 15);
  const int hi   = lane >> 4;

  v8f acc = {};
#pragma unroll
  for (int kb = 0; kb < KB; kb++) {
    v16h a = Aswz[(t * KB + kb) * 32 + lane];
    v16h b = Bswz[(kb * 64 + n) * 2 + hi];
    acc = __builtin_amdgcn_wmma_f32_16x16x32_f16(
        /*neg_a=*/false, a, /*neg_b=*/false, b,
        /*c_mod=*/(short)0, acc, /*reuse_a=*/false, /*reuse_b=*/false);
  }

  const int m0 = blockrow + t * 16 + hi * 8;  // D layout: VGPR r -> M = hi*8 + r
  float bv = 0.0f;
  if (BIAS) bv = bias[n];
#pragma unroll
  for (int r = 0; r < 8; r++) {
    int mm = m0 + r;
    if (!GUARD || mm < Nrows) {
      float v = acc[r] + bv;
      if (RELU) v = fmaxf(v, 0.0f);
      out[(long)mm * HID + n] = v;
    }
  }
}

// ---------- edge scatter: agg[dst] += hw[src] * dinv[src]*dinv[dst] (+ self loops) ----------
__global__ void scatter_kernel(const float* __restrict__ hw, const int* __restrict__ src,
                               const int* __restrict__ dst, const float* __restrict__ dinv,
                               float* __restrict__ agg, int E, int Nn) {
  long idx = (long)blockIdx.x * blockDim.x + threadIdx.x;
  long total = ((long)E + Nn) * HID;
  if (idx >= total) return;
  int  c = (int)(idx & (HID - 1));
  long e = idx >> 6;
  int s, d;
  if (e < (long)E) { s = src[e]; d = dst[e]; }
  else             { s = (int)(e - E); d = s; }   // self loop
  __builtin_prefetch(&hw[(long)s * HID], 0, 1);
  float v = hw[(long)s * HID + c] * dinv[s] * dinv[d];
  atomicAdd(&agg[(long)d * HID + c], v);
}

// ---------- BN stats: stats[0:64]=sum, stats[64:128]=sumsq ----------
__global__ void bn_stats(const float* __restrict__ h, float* __restrict__ stats, int Nn) {
  __shared__ float ssum[HID], ssq[HID];
  if (threadIdx.x < HID) { ssum[threadIdx.x] = 0.f; ssq[threadIdx.x] = 0.f; }
  __syncthreads();
  const int c  = threadIdx.x & (HID - 1);
  const int rp = blockDim.x / HID;
  int r = blockIdx.x * rp + (threadIdx.x >> 6);
  const int rstride = gridDim.x * rp;
  float s = 0.f, q = 0.f;
  for (; r < Nn; r += rstride) {
    float v = h[(long)r * HID + c];
    s += v; q += v * v;
  }
  atomicAdd(&ssum[c], s);
  atomicAdd(&ssq[c], q);
  __syncthreads();
  if (threadIdx.x < HID) {
    atomicAdd(&stats[threadIdx.x], ssum[threadIdx.x]);
    atomicAdd(&stats[HID + threadIdx.x], ssq[threadIdx.x]);
  }
}

__global__ void bn_apply(const float* __restrict__ agg, const float* __restrict__ stats,
                         const float* __restrict__ gamma, const float* __restrict__ beta,
                         float* __restrict__ h, int Nn) {
  long idx = (long)blockIdx.x * blockDim.x + threadIdx.x;
  if (idx >= (long)Nn * HID) return;
  int c = (int)(idx & (HID - 1));
  float invN = 1.0f / (float)Nn;
  float mean = stats[c] * invN;
  float var  = stats[HID + c] * invN - mean * mean;
  float v = (agg[idx] - mean) * rsqrtf(var + BN_EPS) * gamma[c] + beta[c];
  h[idx] = fmaxf(v, 0.0f);
}

// ---------- global mean pool ----------
__global__ void pool_kernel(const float* __restrict__ h, const int* __restrict__ batch,
                            float* __restrict__ sums, float* __restrict__ cnt, int Nn) {
  long idx = (long)blockIdx.x * blockDim.x + threadIdx.x;
  if (idx >= (long)Nn * HID) return;
  int c = (int)(idx & (HID - 1));
  int r = (int)(idx >> 6);
  int g = batch[r];
  atomicAdd(&sums[(long)g * HID + c], h[idx]);
  if (c == 0) atomicAdd(&cnt[g], 1.0f);
}

// ---------- classifier head: one block (64 thr) per graph ----------
__global__ void head_kernel(const float* __restrict__ sums, const float* __restrict__ cnt,
                            const float* __restrict__ W1, const float* __restrict__ b1,
                            const float* __restrict__ W2, const float* __restrict__ b2,
                            float* __restrict__ out) {
  __shared__ float p[HID];
  __shared__ float hc[HID / 2];
  const int g = blockIdx.x;
  const int t = threadIdx.x;
  float cn = fmaxf(cnt[g], 1.0f);
  p[t] = sums[(long)g * HID + t] / cn;
  __syncthreads();
  if (t < HID / 2) {
    float acc = b1[t];
    for (int k = 0; k < HID; k++) acc += p[k] * W1[k * (HID / 2) + t];
    hc[t] = fmaxf(acc, 0.0f);
  }
  __syncthreads();
  if (t < NUM_CLASSES) {
    float acc = b2[t];
    for (int k = 0; k < HID / 2; k++) acc += hc[k] * W2[k * NUM_CLASSES + t];
    out[g * NUM_CLASSES + t] = acc;
  }
}

// host-side helper to launch main grid (guard-free) + optional guarded tail
template <int K, bool BIAS, bool RELU>
static void launch_gemm(const float* A, const _Float16* Wswz, const float* bias, float* out,
                        int Nrows, hipStream_t stream) {
  const int full = Nrows / 32;
  const int rem  = Nrows % 32;
  if (full > 0)
    gemm_wmma<K, BIAS, RELU, false><<<full, dim3(256), 0, stream>>>(A, Wswz, bias, out, Nrows, 0);
  if (rem > 0)
    gemm_wmma<K, BIAS, RELU, true><<<1, dim3(256), 0, stream>>>(A, Wswz, bias, out, Nrows, full * 32);
}

extern "C" void kernel_launch(void* const* d_in, const int* in_sizes, int n_in,
                              void* d_out, int out_size, void* d_ws, size_t ws_size,
                              hipStream_t stream) {
  const float* x     = (const float*)d_in[0];
  const int*   eidx  = (const int*)d_in[1];
  const int*   batch = (const int*)d_in[2];
  const float* W_in  = (const float*)d_in[3];
  const float* b_in  = (const float*)d_in[4];
  const float* Wc    = (const float*)d_in[5];
  const float* bc    = (const float*)d_in[6];   (void)bc; // cancels under training-mode BN
  const float* gamma = (const float*)d_in[7];
  const float* beta  = (const float*)d_in[8];
  const float* W1    = (const float*)d_in[9];
  const float* b1    = (const float*)d_in[10];
  const float* W2    = (const float*)d_in[11];
  const float* b2    = (const float*)d_in[12];
  float* out = (float*)d_out;

  const int IN_DIM = 128;
  const int Nn = in_sizes[0] / IN_DIM;   // 100000
  const int E  = in_sizes[1] / 2;        // 3200000
  const int* src = eidx;
  const int* dst = eidx + E;

  char* ws = (char*)d_ws;
  auto alloc = [&](size_t bytes) -> void* {
    void* p = (void*)ws;
    ws += (bytes + 255) & ~(size_t)255;
    return p;
  };
  float*    deg   = (float*)alloc((size_t)Nn * 4);
  float*    h     = (float*)alloc((size_t)Nn * HID * 4);
  float*    hw    = (float*)alloc((size_t)Nn * HID * 4);
  float*    agg   = (float*)alloc((size_t)Nn * HID * 4);
  _Float16* Wf16  = (_Float16*)alloc((size_t)IN_DIM * HID * 2);
  float*    stats = (float*)alloc(2 * HID * 4);
  float*    psum  = (float*)alloc((size_t)NUM_GRAPHS * HID * 4);
  float*    pcnt  = (float*)alloc((size_t)NUM_GRAPHS * 4);

  const dim3 B(256);
  const long ntotal  = (long)Nn * HID;
  const int  nblocks = (int)((ntotal + 255) / 256);
  const long etotal  = ((long)E + Nn) * HID;
  const int  eblocks = (int)((etotal + 255) / 256);

  // degrees (self loop => init 1) and d^-1/2
  fill_f32<<<(Nn + 255) / 256, B, 0, stream>>>(deg, Nn, 1.0f);
  deg_kernel<<<(E + 255) / 256, B, 0, stream>>>(dst, E, deg);
  rsqrt_kernel<<<(Nn + 255) / 256, B, 0, stream>>>(deg, Nn);

  // input layer: h = relu(x @ W_in + b_in)
  cvt_swz_kernel<<<(IN_DIM * HID + 255) / 256, B, 0, stream>>>(W_in, Wf16, IN_DIM);
  launch_gemm<128, true, true>(x, Wf16, b_in, h, Nn, stream);

  for (int l = 0; l < NUM_LAYERS; l++) {
    cvt_swz_kernel<<<(HID * HID + 255) / 256, B, 0, stream>>>(Wc + (size_t)l * HID * HID, Wf16, HID);
    launch_gemm<HID, false, false>(h, Wf16, nullptr, hw, Nn, stream);
    fill_f32<<<nblocks, B, 0, stream>>>(agg, (int)ntotal, 0.0f);
    scatter_kernel<<<eblocks, B, 0, stream>>>(hw, src, dst, deg, agg, E, Nn);
    fill_f32<<<1, B, 0, stream>>>(stats, 2 * HID, 0.0f);
    bn_stats<<<512, B, 0, stream>>>(agg, stats, Nn);
    bn_apply<<<nblocks, B, 0, stream>>>(agg, stats, gamma + l * HID, beta + l * HID, h, Nn);
  }

  // mean pool + head
  fill_f32<<<(NUM_GRAPHS * HID + 255) / 256, B, 0, stream>>>(psum, NUM_GRAPHS * HID, 0.0f);
  fill_f32<<<(NUM_GRAPHS + 255) / 256, B, 0, stream>>>(pcnt, NUM_GRAPHS, 0.0f);
  pool_kernel<<<nblocks, B, 0, stream>>>(h, batch, psum, pcnt, Nn);
  head_kernel<<<NUM_GRAPHS, dim3(HID), 0, stream>>>(psum, pcnt, W1, b1, W2, b2, out);
}
